// LlamaSdpaAttention_24369644437504
// MI455X (gfx1250) — matrix operand order repaired and checked by reading, not compiled, not run on hardware
//
#include <hip/hip_runtime.h>
#include <hip/hip_bf16.h>

typedef __attribute__((ext_vector_type(16))) __bf16 v16bf;
typedef __attribute__((ext_vector_type(8)))  float  v8f;

constexpr int T_SEQ = 4096, HID = 2048, DH = 128, NQ = 16, NKV = 4;

union Frag {
  unsigned int u[8];
  uint4 q[2];
  v16bf v;
};

__device__ __forceinline__ unsigned short f2bf(float f) {
  unsigned int u = __builtin_bit_cast(unsigned int, f);
  u += 0x7FFFu + ((u >> 16) & 1u);          // round-to-nearest-even
  return (unsigned short)(u >> 16);
}
__device__ __forceinline__ unsigned int pk2(float a, float b) {
  return (unsigned int)f2bf(a) | ((unsigned int)f2bf(b) << 16);
}
__device__ __forceinline__ float bf2f(unsigned short u) {
  return __builtin_bit_cast(float, (unsigned int)u << 16);
}

// --- single-instruction lane XOR exchange via ds_swizzle (group-of-32 mode) --
template <int MASK>
__device__ __forceinline__ float swz_xor(float x) {
  return __builtin_bit_cast(
      float, __builtin_amdgcn_ds_swizzle(__builtin_bit_cast(int, x), (MASK << 10) | 0x1F));
}
__device__ __forceinline__ float half_reduce_max(float x) {  // over 16-lane halves
  x = fmaxf(x, swz_xor<1>(x));
  x = fmaxf(x, swz_xor<2>(x));
  x = fmaxf(x, swz_xor<4>(x));
  x = fmaxf(x, swz_xor<8>(x));
  return x;
}
__device__ __forceinline__ float half_reduce_sum(float x) {
  x += swz_xor<1>(x);
  x += swz_xor<2>(x);
  x += swz_xor<4>(x);
  x += swz_xor<8>(x);
  return x;
}

// --- CDNA5 async global->LDS copy (ASYNCcnt) --------------------------------
__device__ __forceinline__ void async16(const unsigned short* g, const unsigned short* l) {
  unsigned int lds_off = (unsigned int)(unsigned long long)l;  // generic ptr low 32b = LDS offset
  asm volatile("global_load_async_to_lds_b128 %0, %1, off"
               :: "v"(lds_off), "v"(g) : "memory");
}
__device__ __forceinline__ void wait_async0() {
  asm volatile("s_wait_asynccnt 0x0" ::: "memory");
}

// ---------------------------------------------------------------------------
// fp32 -> bf16 bulk convert (done once per tensor)
// ---------------------------------------------------------------------------
__global__ __launch_bounds__(256) void cvt_bf16(const float* __restrict__ src,
                                                unsigned short* __restrict__ dst, int n4) {
  int i = blockIdx.x * 256 + threadIdx.x;
  if (i >= n4) return;
  float4 f = ((const float4*)src)[i];
  uint2 o;
  o.x = pk2(f.x, f.y);
  o.y = pk2(f.z, f.w);
  ((uint2*)dst)[i] = o;
}

// ---------------------------------------------------------------------------
// GEMM: C[M,N] = X[M,K] * W[N,K]^T   (bf16 in via async LDS, f32 acc)
// 256 thr = 8 waves, 128x128 tile, wave 32x64, K-step 32, double-buffered LDS.
// ---------------------------------------------------------------------------
#define GSTRIDE 40   // padded ushort row stride (80B, 16B aligned)

template <bool OUT_F32>
__global__ __launch_bounds__(256) void gemm_bf16(const unsigned short* __restrict__ X,
                                                 const unsigned short* __restrict__ W,
                                                 void* __restrict__ Cout,
                                                 int M, int N, int K) {
  __shared__ __align__(16) unsigned short As[2][128 * GSTRIDE];
  __shared__ __align__(16) unsigned short Bs[2][128 * GSTRIDE];
  const int tid  = threadIdx.x;
  const int wave = tid >> 5, lane = tid & 31;
  const int l16  = lane & 15;
  const int hi   = (lane >> 4) & 1;
  const int bm = blockIdx.x * 128, bn = blockIdx.y * 128;
  const int wm = (wave & 3) * 32,  wn = (wave >> 2) * 64;

  v8f acc[2][4];
#pragma unroll
  for (int i = 0; i < 2; i++)
#pragma unroll
    for (int j = 0; j < 4; j++) acc[i][j] = (v8f){0.f,0.f,0.f,0.f,0.f,0.f,0.f,0.f};

  auto stage = [&](int buf, int k0) {
#pragma unroll
    for (int c = tid; c < 512; c += 256) {
      int r = c >> 2, cc = (c & 3) * 8;
      async16(X + (size_t)(bm + r) * K + k0 + cc, &As[buf][r * GSTRIDE + cc]);
      async16(W + (size_t)(bn + r) * K + k0 + cc, &Bs[buf][r * GSTRIDE + cc]);
    }
  };

  stage(0, 0);
  wait_async0();
  __syncthreads();

  const int nk = K >> 5;
  for (int ik = 0; ik < nk; ik++) {
    const int cur = ik & 1;
    if (ik + 1 < nk) stage(cur ^ 1, (ik + 1) << 5);

    Frag af[2];
#pragma unroll
    for (int mt = 0; mt < 2; mt++) {
      const unsigned short* ar = &As[cur][(wm + mt * 16 + l16) * GSTRIDE + (hi ? 8 : 0)];
      af[mt].q[0] = *(const uint4*)ar;
      af[mt].q[1] = *(const uint4*)(ar + 16);
    }
#pragma unroll
    for (int nt = 0; nt < 4; nt++) {
      Frag bfg;
      const unsigned short* br = &Bs[cur][(wn + nt * 16 + l16) * GSTRIDE + (hi ? 8 : 0)];
      bfg.q[0] = *(const uint4*)br;
      bfg.q[1] = *(const uint4*)(br + 16);
#pragma unroll
      for (int mt = 0; mt < 2; mt++)
        acc[mt][nt] = __builtin_amdgcn_wmma_f32_16x16x32_bf16(
            false, af[mt].v, false, bfg.v, (short)0, acc[mt][nt], false, false);
    }
    wait_async0();
    __syncthreads();
  }

#pragma unroll
  for (int mt = 0; mt < 2; mt++)
#pragma unroll
    for (int nt = 0; nt < 4; nt++)
#pragma unroll
      for (int j = 0; j < 8; j++) {
        size_t idx = (size_t)(bm + wm + mt * 16 + j + hi * 8) * N + (bn + wn + nt * 16 + l16);
        if constexpr (OUT_F32) ((float*)Cout)[idx] = acc[mt][nt][j];
        else                   ((unsigned short*)Cout)[idx] = f2bf(acc[mt][nt][j]);
      }
}

// ---------------------------------------------------------------------------
// RoPE in place on bf16 x[T, H, DH]
// ---------------------------------------------------------------------------
__global__ __launch_bounds__(256) void rope_bf(unsigned short* __restrict__ x,
                                               const int* __restrict__ pos, int H) {
  const int half = DH / 2;
  int idx = blockIdx.x * blockDim.x + threadIdx.x;
  int total = T_SEQ * H * half;
  if (idx >= total) return;
  int i = idx % half;
  int h = (idx / half) % H;
  int t = idx / (half * H);
  float p   = (float)pos[t];
  float inv = __powf(10000.0f, -(float)i / (float)half);
  float f = p * inv;
  float c = __cosf(f), s = __sinf(f);
  unsigned short* b = x + ((size_t)t * H + h) * DH;
  float x1 = bf2f(b[i]), x2 = bf2f(b[i + half]);
  b[i]        = f2bf(x1 * c - x2 * s);
  b[i + half] = f2bf(x2 * c + x1 * s);
}

// ---------------------------------------------------------------------------
// V transpose: vt[kvh][d][t] = v[t][kvh][d]   (bf16, LDS-tiled 64x64)
// ---------------------------------------------------------------------------
__global__ __launch_bounds__(256) void transpose_v(const unsigned short* __restrict__ vb,
                                                   unsigned short* __restrict__ vt) {
  __shared__ unsigned short tile[64][65];
  const int t0 = blockIdx.x * 64, d0 = blockIdx.y * 64, kvh = blockIdx.z;
  const int tx = threadIdx.x & 63, ty = threadIdx.x >> 6;
#pragma unroll
  for (int i = ty; i < 64; i += 4)
    tile[i][tx] = vb[((size_t)(t0 + i) * NKV + kvh) * DH + d0 + tx];
  __syncthreads();
#pragma unroll
  for (int i = ty; i < 64; i += 4)
    vt[((size_t)kvh * DH + d0 + i) * T_SEQ + t0 + tx] = tile[tx][i];
}

// ---------------------------------------------------------------------------
// Flash attention: grid (T/64, NQ), 128 thr (4 waves), 16 q-rows/wave,
// 32-key blocks, async double-buffered K/V staging, bf16 WMMA, f32 online softmax.
// ---------------------------------------------------------------------------
#define KSTRIDE 136  // 272B row (16B aligned)
#define VSTRIDE 40   // 80B row  (16B aligned)

__global__ __launch_bounds__(128) void flash_k(const unsigned short* __restrict__ Qm,
                                               const unsigned short* __restrict__ Km,
                                               const unsigned short* __restrict__ Vt,
                                               unsigned short* __restrict__ Om) {
  __shared__ __align__(16) unsigned short Ks[2][32 * KSTRIDE];     // [key][d]
  __shared__ __align__(16) unsigned short Vs[2][128 * VSTRIDE];    // [d][key]
  __shared__ __align__(16) unsigned short Ps[4][16 * VSTRIDE];     // per-wave P tile
  const int tid  = threadIdx.x;
  const int wave = tid >> 5, lane = tid & 31;
  const int l16  = lane & 15;
  const int hi   = (lane >> 4) & 1;
  const int qblock = blockIdx.x * 64;
  const int h   = blockIdx.y;
  const int kvh = h / (NQ / NKV);
  const int qrow0 = qblock + wave * 16;
  const float scale = 0.08838834764831845f;  // 1/sqrt(128)

  // Q fragments: direct packed bf16 loads (A layout, 4 K-chunks over DH)
  Frag qf[4];
  {
    const unsigned short* qp = Qm + ((size_t)(qrow0 + l16) * NQ + h) * DH;
#pragma unroll
    for (int c = 0; c < 4; c++) {
      int d0 = c * 32 + (hi ? 8 : 0);
      qf[c].q[0] = *(const uint4*)(qp + d0);
      qf[c].q[1] = *(const uint4*)(qp + d0 + 16);
    }
  }

  auto stage = [&](int buf, int kb) {
#pragma unroll
    for (int c = tid; c < 512; c += 128) {          // K: 32 rows x 16 chunks
      int key = c >> 4, cc = (c & 15) * 8;
      async16(Km + ((size_t)(kb + key) * NKV + kvh) * DH + cc,
              &Ks[buf][key * KSTRIDE + cc]);
    }
#pragma unroll
    for (int c = tid; c < 512; c += 128) {          // V^T: 128 rows x 4 chunks
      int d = c >> 2, cc = (c & 3) * 8;
      async16(Vt + ((size_t)kvh * DH + d) * T_SEQ + kb + cc,
              &Vs[buf][d * VSTRIDE + cc]);
    }
  };

  float m_r[8], l_r[8];
#pragma unroll
  for (int j = 0; j < 8; j++) { m_r[j] = -3.0e38f; l_r[j] = 0.f; }
  v8f o[8];
#pragma unroll
  for (int i = 0; i < 8; i++) o[i] = (v8f){0.f,0.f,0.f,0.f,0.f,0.f,0.f,0.f};

  const int myM = hi ? 8 : 0;
  const int nblocks = (qblock >> 5) + 2;

  stage(0, 0);
  wait_async0();
  __syncthreads();

  for (int ib = 0; ib < nblocks; ib++) {
    const int kb  = ib << 5;
    const int cur = ib & 1;
    if (ib + 1 < nblocks) stage(cur ^ 1, kb + 32);

    // S = Q * K^T over DH=128 (two 16-wide key tiles)
    v8f s0 = (v8f){0.f,0.f,0.f,0.f,0.f,0.f,0.f,0.f};
    v8f s1 = (v8f){0.f,0.f,0.f,0.f,0.f,0.f,0.f,0.f};
#pragma unroll
    for (int c = 0; c < 4; c++) {
      Frag b0, b1;
      const unsigned short* kr0 = &Ks[cur][l16 * KSTRIDE + c * 32 + (hi ? 8 : 0)];
      const unsigned short* kr1 = kr0 + 16 * KSTRIDE;
      b0.q[0] = *(const uint4*)kr0; b0.q[1] = *(const uint4*)(kr0 + 16);
      b1.q[0] = *(const uint4*)kr1; b1.q[1] = *(const uint4*)(kr1 + 16);
      s0 = __builtin_amdgcn_wmma_f32_16x16x32_bf16(false, qf[c].v, false, b0.v,
                                                   (short)0, s0, false, false);
      s1 = __builtin_amdgcn_wmma_f32_16x16x32_bf16(false, qf[c].v, false, b1.v,
                                                   (short)0, s1, false, false);
    }

    // scale + causal mask + online softmax (rows live across 16-lane halves)
#pragma unroll
    for (int j = 0; j < 8; j++) {
      int qr = qrow0 + myM + j;
      float v0 = (kb + l16      <= qr) ? s0[j] * scale : -3.0e38f;
      float v1 = (kb + 16 + l16 <= qr) ? s1[j] * scale : -3.0e38f;
      float mx = half_reduce_max(fmaxf(v0, v1));
      float mn    = fmaxf(m_r[j], mx);
      float alpha = __expf(m_r[j] - mn);
      float p0 = __expf(v0 - mn);
      float p1 = __expf(v1 - mn);
      float rs = half_reduce_sum(p0 + p1);
      l_r[j] = l_r[j] * alpha + rs;
      m_r[j] = mn;
      Ps[wave][(myM + j) * VSTRIDE + l16]      = f2bf(p0);
      Ps[wave][(myM + j) * VSTRIDE + 16 + l16] = f2bf(p1);
#pragma unroll
      for (int dt = 0; dt < 8; dt++) o[dt][j] *= alpha;
    }

    // P back as A-layout bf16 (same-wave DS ops in order: no barrier needed)
    Frag pf;
    const unsigned short* pr = &Ps[wave][l16 * VSTRIDE + (hi ? 8 : 0)];
    pf.q[0] = *(const uint4*)pr;
    pf.q[1] = *(const uint4*)(pr + 16);

    // O += P * V
#pragma unroll
    for (int dt = 0; dt < 8; dt++) {
      Frag bv;
      const unsigned short* vr = &Vs[cur][(dt * 16 + l16) * VSTRIDE + (hi ? 8 : 0)];
      bv.q[0] = *(const uint4*)vr; bv.q[1] = *(const uint4*)(vr + 16);
      o[dt] = __builtin_amdgcn_wmma_f32_16x16x32_bf16(false, pf.v, false, bv.v,
                                                      (short)0, o[dt], false, false);
    }

    wait_async0();
    __syncthreads();
  }

#pragma unroll
  for (int j = 0; j < 8; j++) {
    float invl = 1.0f / l_r[j];
    size_t row = (size_t)(qrow0 + myM + j);
#pragma unroll
    for (int dt = 0; dt < 8; dt++)
      Om[row * HID + h * DH + dt * 16 + l16] = f2bf(o[dt][j] * invl);
  }
}

// ---------------------------------------------------------------------------
extern "C" void kernel_launch(void* const* d_in, const int* in_sizes, int n_in,
                              void* d_out, int out_size, void* d_ws, size_t ws_size,
                              hipStream_t stream) {
  const float* hs = (const float*)d_in[0];
  const float* wq = (const float*)d_in[1];
  const float* wk = (const float*)d_in[2];
  const float* wv = (const float*)d_in[3];
  const float* wo = (const float*)d_in[4];
  const int*  pos = (const int*)d_in[5];
  float* out = (float*)d_out;

  unsigned short* p = (unsigned short*)d_ws;
  unsigned short* hs_bf = p; p += (size_t)T_SEQ * HID;
  unsigned short* wq_bf = p; p += (size_t)HID * HID;
  unsigned short* wk_bf = p; p += (size_t)NKV * DH * HID;
  unsigned short* wv_bf = p; p += (size_t)NKV * DH * HID;
  unsigned short* wo_bf = p; p += (size_t)HID * HID;
  unsigned short* q_bf  = p; p += (size_t)T_SEQ * NQ * DH;
  unsigned short* k_bf  = p; p += (size_t)T_SEQ * NKV * DH;
  unsigned short* v_bf  = p; p += (size_t)T_SEQ * NKV * DH;
  unsigned short* vt_bf = p; p += (size_t)T_SEQ * NKV * DH;
  unsigned short* at_bf = p;

  // one-time fp32 -> bf16 conversions
  cvt_bf16<<<(T_SEQ * HID / 4 + 255) / 256, 256, 0, stream>>>(hs, hs_bf, T_SEQ * HID / 4);
  cvt_bf16<<<(HID * HID / 4 + 255) / 256, 256, 0, stream>>>(wq, wq_bf, HID * HID / 4);
  cvt_bf16<<<(NKV * DH * HID / 4 + 255) / 256, 256, 0, stream>>>(wk, wk_bf, NKV * DH * HID / 4);
  cvt_bf16<<<(NKV * DH * HID / 4 + 255) / 256, 256, 0, stream>>>(wv, wv_bf, NKV * DH * HID / 4);
  cvt_bf16<<<(HID * HID / 4 + 255) / 256, 256, 0, stream>>>(wo, wo_bf, HID * HID / 4);

  // QKV projections
  gemm_bf16<false><<<dim3(T_SEQ / 128, HID / 128), 256, 0, stream>>>(hs_bf, wq_bf, q_bf, T_SEQ, HID, HID);
  gemm_bf16<false><<<dim3(T_SEQ / 128, (NKV * DH) / 128), 256, 0, stream>>>(hs_bf, wk_bf, k_bf, T_SEQ, NKV * DH, HID);
  gemm_bf16<false><<<dim3(T_SEQ / 128, (NKV * DH) / 128), 256, 0, stream>>>(hs_bf, wv_bf, v_bf, T_SEQ, NKV * DH, HID);

  // RoPE
  rope_bf<<<(T_SEQ * NQ  * (DH / 2) + 255) / 256, 256, 0, stream>>>(q_bf, pos, NQ);
  rope_bf<<<(T_SEQ * NKV * (DH / 2) + 255) / 256, 256, 0, stream>>>(k_bf, pos, NKV);

  // V^T for flash B-fragments
  transpose_v<<<dim3(T_SEQ / 64, DH / 64, NKV), 256, 0, stream>>>(v_bf, vt_bf);

  // Causal GQA flash attention
  flash_k<<<dim3(T_SEQ / 64, NQ), 128, 0, stream>>>(q_bf, k_bf, vt_bf, at_bf);

  // Output projection (fp32 out)
  gemm_bf16<true><<<dim3(T_SEQ / 128, HID / 128), 256, 0, stream>>>(at_bf, wo_bf, out, T_SEQ, HID, HID);
}
